// Attention_40037685134082
// MI455X (gfx1250) — matrix operand order repaired and checked by reading, not compiled, not run on hardware
//
#include <hip/hip_runtime.h>
#include <hip/hip_bf16.h>

// ---------------------------------------------------------------------------
// Sigmoid attention for MI455X (gfx1250), bf16 WMMA pipeline.
//   B=2, N=2048, C=1024, H=16, Dh=64
//   attn = sigmoid(q*Dh^-0.5 @ k^T * exp(0) + bias);  out = attn @ v;  proj.
// No softmax -> single streaming pass over keys (fully fused attention).
// The 537 MB fp32 attn-matrix output stream uses non-temporal stores so it
// does not evict the resident K/V and weight tiles from the 192 MB L2.
// ---------------------------------------------------------------------------

#define BB   2
#define NN   2048
#define CC   1024
#define HH   16
#define DH   64
#define ATTN_BIAS (-7.625f)
#define QK_SCALE  (0.125f)     /* Dh^-0.5, exp(ATTN_TEMP)=1 */

typedef __attribute__((ext_vector_type(16))) __bf16 v16bf;
typedef __attribute__((ext_vector_type(8)))  __bf16 v8bf;
typedef __attribute__((ext_vector_type(8)))  float  v8f;

// A-fragment (16x32 bf16): lane L holds row M=L%16; with g=L/16 the
// 16 halves cover K = g*16 + j (contiguous, matching the B-fragments).
// `p` must already include row*lda + k0 + g*16.
static __device__ inline v16bf afrag(const __bf16* p) {
  v8bf lo = *(const v8bf*)p;
  v8bf hi = *(const v8bf*)(p + 8);
  return __builtin_shufflevector(lo, hi, 0,1,2,3,4,5,6,7,8,9,10,11,12,13,14,15);
}

static __device__ inline v8f wmma_bf16(v16bf a, v16bf b, v8f c) {
  // emits v_wmma_f32_16x16x32_bf16
  return __builtin_amdgcn_wmma_f32_16x16x32_bf16(false, a, false, b,
                                                 (short)0, c, false, false);
}

// ---------------------------------------------------------------------------
__global__ void cvt_f32_bf16(const float* __restrict__ src,
                             __bf16* __restrict__ dst, int n) {
  int i = blockIdx.x * blockDim.x + threadIdx.x;
  if (i < n) dst[i] = (__bf16)src[i];
}

// vt[b][h][d][m] = v[b][m][h][d]   (v = slice 2 of qkv3 [B,N,3,H,Dh])
__global__ void v_transpose(const __bf16* __restrict__ qkv3,
                            __bf16* __restrict__ vt) {
  int idx = blockIdx.x * blockDim.x + threadIdx.x;     // B*H*DH*NN = 2^22
  int m = idx & (NN - 1);
  int d = (idx >> 11) & (DH - 1);
  int h = (idx >> 17) & (HH - 1);
  int b = idx >> 21;
  vt[idx] = qkv3[((size_t)(b * NN + m) * 3 + 2) * CC + h * DH + d];
}

// ---------------------------------------------------------------------------
// D[M,Nout] = A[M,K](bf16, row-major) @ W[Nout,K](bf16, row-major)^T
// One wave computes a 16x64 tile (A-fragment reused across 4 B-fragments).
// NT_OUT: fp32 result is a final output (never re-read) -> non-temporal store.
__global__ void gemm_bf16(const __bf16* __restrict__ A,
                          const __bf16* __restrict__ W,
                          __bf16* __restrict__ Dbf, float* __restrict__ Df,
                          int M, int Nout, int K) {
  const int lane = threadIdx.x & 31;
  const int wid  = threadIdx.x >> 5;
  const int g    = lane >> 4;
  const int ln   = lane & 15;
  const int tilesN = Nout >> 6;
  const int wflat = blockIdx.x * (blockDim.x >> 5) + wid;
  if (wflat >= (M >> 4) * tilesN) return;
  const int mbase = (wflat / tilesN) << 4;
  const int nbase = (wflat % tilesN) << 6;

  const __bf16* arow = A + (size_t)(mbase + ln) * K + g * 16;
  const __bf16* wrow = W + (size_t)(nbase + ln) * K + g * 16;

  v8f acc0 = {}, acc1 = {}, acc2 = {}, acc3 = {};
  for (int k0 = 0; k0 < K; k0 += 32) {
    v16bf a  = afrag(arow + k0);
    v16bf w0 = *(const v16bf*)(wrow + k0);
    v16bf w1 = *(const v16bf*)(wrow + (size_t)16 * K + k0);
    v16bf w2 = *(const v16bf*)(wrow + (size_t)32 * K + k0);
    v16bf w3 = *(const v16bf*)(wrow + (size_t)48 * K + k0);
    acc0 = wmma_bf16(a, w0, acc0);
    acc1 = wmma_bf16(a, w1, acc1);
    acc2 = wmma_bf16(a, w2, acc2);
    acc3 = wmma_bf16(a, w3, acc3);
  }
#pragma unroll
  for (int t = 0; t < 4; ++t) {
    v8f acc = (t == 0) ? acc0 : (t == 1) ? acc1 : (t == 2) ? acc2 : acc3;
    int col = nbase + t * 16 + ln;
#pragma unroll
    for (int r = 0; r < 8; ++r) {
      size_t idx = (size_t)(mbase + r + 8 * g) * Nout + col;
      if (Df)  __builtin_nontemporal_store(acc[r], &Df[idx]);  // final output
      if (Dbf) Dbf[idx] = (__bf16)acc[r];                      // re-read: RT
    }
  }
}

// ---------------------------------------------------------------------------
// Fused sigmoid attention.  Block = 128 threads = 4 waves; each wave owns 16
// query rows (block covers 64).  Streams over keys in 32-wide tiles:
//   S(16x32) via 4 wmma -> sigmoid -> fp32 attn_matrix NT-store + bf16 LDS
//   tile -> LDS re-swizzle to A-fragment -> 4 wmma accumulate into O(16x64).
__global__ void attn_fused(const __bf16* __restrict__ qkv3,
                           const __bf16* __restrict__ vt,
                           float* __restrict__ attn_out,
                           float* __restrict__ av_out,
                           __bf16* __restrict__ avb) {
  __shared__ __align__(32) __bf16 lds[4][16 * 32];

  const int lane = threadIdx.x & 31;
  const int wid  = threadIdx.x >> 5;
  const int g    = lane >> 4;
  const int ln   = lane & 15;

  const int qblocks = NN / 64;
  const int bh   = blockIdx.x / qblocks;
  const int qblk = blockIdx.x % qblocks;
  const int b = bh / HH, h = bh % HH;
  const int qbase = qblk * 64 + wid * 16;

  // Q fragments (resident for the whole key loop), K-dim = 64 -> 2 chunks.
  const __bf16* qrowp =
      qkv3 + (size_t)(b * NN + qbase + ln) * 3 * CC + h * DH + g * 16;
  const v16bf qf0 = afrag(qrowp);
  const v16bf qf1 = afrag(qrowp + 32);

  const __bf16* krows = qkv3 + (size_t)b * NN * 3 * CC + CC + h * DH;
  const __bf16* vrows = vt + (size_t)(b * HH + h) * DH * NN + (size_t)ln * NN;
  float* attn_bh = attn_out + (size_t)(b * HH + h) * NN * NN;

  v8f o0 = {}, o1 = {}, o2 = {}, o3 = {};

  for (int m0 = 0; m0 < NN; m0 += 32) {
    // Prefetch next key/value tiles (global_prefetch_b8) while we compute.
    if (m0 + 32 < NN) {
      __builtin_prefetch(krows + (size_t)(m0 + 32 + ln) * 3 * CC + g * 16, 0, 3);
      __builtin_prefetch(vrows + m0 + 32 + g * 16, 0, 3);
    }
#pragma unroll
    for (int s = 0; s < 2; ++s) {
      const int key16 = m0 + s * 16;
      // K^T B-fragment: lane ln = key column, contiguous Dh run per lane.
      const __bf16* kp = krows + (size_t)(key16 + ln) * 3 * CC + g * 16;
      v16bf kf0 = *(const v16bf*)kp;
      v16bf kf1 = *(const v16bf*)(kp + 32);
      v8f sacc = {};
      sacc = wmma_bf16(qf0, kf0, sacc);
      sacc = wmma_bf16(qf1, kf1, sacc);
#pragma unroll
      for (int r = 0; r < 8; ++r) {
        float sv = sacc[r] * QK_SCALE + ATTN_BIAS;
        float p  = 1.0f / (1.0f + __expf(-sv));
        int   qr = r + 8 * g;
        // 537MB write-once stream: non-temporal so it doesn't wash out L2.
        __builtin_nontemporal_store(
            p, &attn_bh[(size_t)(qbase + qr) * NN + key16 + ln]);
        lds[wid][qr * 32 + s * 16 + ln] = (__bf16)p;           // bf16 staging
      }
    }
    // per-wave LDS write -> cross-lane read ordering
    asm volatile("s_wait_dscnt 0" ::: "memory");

    // P A-fragment (16 queries x 32 keys) out of LDS
    const __bf16* lp = &lds[wid][ln * 32 + g * 16];
    v8bf plo = *(const v8bf*)lp;
    v8bf phi = *(const v8bf*)(lp + 8);
    v16bf pf = __builtin_shufflevector(plo, phi,
                 0,1,2,3,4,5,6,7,8,9,10,11,12,13,14,15);

    // V B-fragments: vt is [B,H,Dh,N] so keys are contiguous per lane.
    const __bf16* vp = vrows + m0 + g * 16;
    o0 = wmma_bf16(pf, *(const v16bf*)(vp),                    o0);
    o1 = wmma_bf16(pf, *(const v16bf*)(vp + (size_t)16 * NN),  o1);
    o2 = wmma_bf16(pf, *(const v16bf*)(vp + (size_t)32 * NN),  o2);
    o3 = wmma_bf16(pf, *(const v16bf*)(vp + (size_t)48 * NN),  o3);

    asm volatile("s_wait_dscnt 0" ::: "memory");  // loads done before reuse
  }

  // attn @ v  ->  [B,N,C] fp32 output (final, NT) + bf16 copy for projection
#pragma unroll
  for (int t = 0; t < 4; ++t) {
    v8f o = (t == 0) ? o0 : (t == 1) ? o1 : (t == 2) ? o2 : o3;
#pragma unroll
    for (int r = 0; r < 8; ++r) {
      int row = qbase + r + 8 * g;
      size_t idx = (size_t)(b * NN + row) * CC + h * DH + t * 16 + ln;
      float val = o[r];
      __builtin_nontemporal_store(val, &av_out[idx]);
      avb[idx] = (__bf16)val;                       // re-read by proj GEMM
    }
  }
}

// ---------------------------------------------------------------------------
extern "C" void kernel_launch(void* const* d_in, const int* in_sizes, int n_in,
                              void* d_out, int out_size, void* d_ws, size_t ws_size,
                              hipStream_t stream) {
  const float* x      = (const float*)d_in[0];   // [B,N,C]
  const float* w_qkv  = (const float*)d_in[1];   // [3C,C]
  const float* w_proj = (const float*)d_in[2];   // [C,C]
  float* out = (float*)d_out;

  const size_t nX    = (size_t)BB * NN * CC;        // 4,194,304
  const size_t nWqkv = (size_t)3 * CC * CC;         // 3,145,728
  const size_t nWprj = (size_t)CC * CC;             // 1,048,576
  const size_t nQKV  = (size_t)BB * NN * 3 * CC;    // 12,582,912

  char* ws = (char*)d_ws;
  size_t off = 0;
  __bf16* xb     = (__bf16*)(ws + off); off += nX    * 2;
  __bf16* wqkvb  = (__bf16*)(ws + off); off += nWqkv * 2;
  __bf16* wprojb = (__bf16*)(ws + off); off += nWprj * 2;
  __bf16* qkv3b  = (__bf16*)(ws + off); off += nQKV  * 2;  // [B,N,3,H,Dh]
  __bf16* vtb    = (__bf16*)(ws + off); off += nX    * 2;  // [B,H,Dh,N]
  __bf16* avb    = (__bf16*)(ws + off); off += nX    * 2;  // [B,N,C]

  float* attn_out = out;                                    // [B,H,N,N]
  float* av_out   = out + (size_t)BB * HH * NN * NN;        // [B,N,C]
  float* proj_out = av_out + nX;                            // [B,N,C]

  // 1) fp32 -> bf16 staging
  cvt_f32_bf16<<<(int)(nX    / 256), 256, 0, stream>>>(x,      xb,     (int)nX);
  cvt_f32_bf16<<<(int)(nWqkv / 256), 256, 0, stream>>>(w_qkv,  wqkvb,  (int)nWqkv);
  cvt_f32_bf16<<<(int)(nWprj / 256), 256, 0, stream>>>(w_proj, wprojb, (int)nWprj);

  // 2) QKV projection: [4096,1024] @ [3072,1024]^T -> bf16 [B,N,3,H,Dh]
  {
    int M = BB * NN, Nout = 3 * CC, K = CC;
    int waves = (M / 16) * (Nout / 64);               // 12288
    gemm_bf16<<<waves / 8, 256, 0, stream>>>(xb, wqkvb, qkv3b, nullptr,
                                             M, Nout, K);
  }

  // 3) V transpose to [B,H,Dh,N] for contiguous WMMA B-fragments
  v_transpose<<<(int)(nX / 256), 256, 0, stream>>>(qkv3b, vtb);

  // 4) fused sigmoid attention (attn_matrix fp32 + attn@v fp32 + bf16 copy)
  attn_fused<<<BB * HH * (NN / 64), 128, 0, stream>>>(qkv3b, vtb,
                                                      attn_out, av_out, avb);

  // 5) output projection: [4096,1024] @ [1024,1024]^T -> fp32
  {
    int M = BB * NN, Nout = CC, K = CC;
    int waves = (M / 16) * (Nout / 64);               // 4096
    gemm_bf16<<<waves / 8, 256, 0, stream>>>(avb, wprojb, nullptr, proj_out,
                                             M, Nout, K);
  }
}
